// HFOpenMoeDecoderLayer_64089501991567
// MI455X (gfx1250) — matrix operand order repaired and checked by reading, not compiled, not run on hardware
//
#include <hip/hip_runtime.h>
#include <hip/hip_bf16.h>

// ---------------------------------------------------------------------------
// CDNA5 (gfx1250) decoder layer: wave32, WMMA f32_16x16x32_f16 everywhere,
// async global->LDS staging for the f16 A operand (ASYNCcnt path).
// ---------------------------------------------------------------------------

typedef __attribute__((ext_vector_type(16))) _Float16 v16h;
typedef __attribute__((ext_vector_type(8)))  _Float16 v8h;
typedef __attribute__((ext_vector_type(8)))  float    v8f;

union F16x16 { v16h v; v8h h[2]; };

static __device__ __forceinline__ v8f wmma_f16(const F16x16& a, const F16x16& b, v8f c) {
  return __builtin_amdgcn_wmma_f32_16x16x32_f16(false, a.v, false, b.v,
                                                (short)0, c, false, false);
}

// LDS byte offset of a __shared__ object (addrspace(3) pointers are 32-bit
// offsets from the wave's LDS base, which is what async-to-LDS ops consume).
static __device__ __forceinline__ unsigned lds_off(const void* p) {
  return (unsigned)(unsigned long long)(__attribute__((address_space(3))) const char*)p;
}

// GLOBAL_LOAD_ASYNC_TO_LDS_B128: copies 16B global->LDS, offset added to both addresses,
// tracked with ASYNCcnt. GV addressing (64-bit vaddr, saddr=off).
static __device__ __forceinline__ void async_g2l_b128(unsigned ldsoff,
                                                      const void* gaddr) {
  unsigned long long ga = (unsigned long long)gaddr;
  asm volatile("global_load_async_to_lds_b128 %0, %1, off"
               :: "v"(ldsoff), "v"(ga) : "memory");
  asm volatile("global_load_async_to_lds_b128 %0, %1, off offset:16"
               :: "v"(ldsoff), "v"(ga) : "memory");
}
static __device__ __forceinline__ void wait_async0() {
  asm volatile("s_wait_asynccnt 0x0" ::: "memory");
}

#define NTOK 4096          // B*S
#define DIM  1024
#define SEQ  2048
#define NH   16
#define HD   64
#define NE   16
#define CAPX 320           // int(1.25*4096/16)
#define ECAP (NE*CAPX)     // 5120
#define FFN2 4096          // 2*FFN
#define FFN1 2048

// ---------------------------------------------------------------------------
// Tiled GEMM:  C[M,N](f32) (+= Res) = A[M,K](f16) * B[K,N](f32->f16)
// Block tile 64x128, K-step 32, 128 threads = 4 waves (2x2), wave tile 32x64.
// A staged via async global->LDS (no VGPR round-trip); B converted fp32->f16
// in registers and staged transposed ([n][k]) so B-fragment loads are
// contiguous ds_load_b128 per the ISA B-matrix lane layout.
// ---------------------------------------------------------------------------
template<bool ADD_RES>
__global__ __launch_bounds__(128)
void gemm_wmma(const _Float16* __restrict__ A, int lda, long strideA,
               const float*    __restrict__ B, int ldb, long strideB,
               float*          __restrict__ C, int ldc, long strideC,
               const float*    __restrict__ Res, int K)
{
  __shared__ __align__(16) _Float16 As[64 * 32];    // [m][k]
  __shared__ __align__(16) _Float16 Bs[128 * 32];   // [n][k]  (transposed)

  const int e  = blockIdx.z;
  A += (long)e * strideA;  B += (long)e * strideB;  C += (long)e * strideC;
  const int m0 = blockIdx.y * 64;
  const int n0 = blockIdx.x * 128;
  const int tid  = threadIdx.x;
  const int lane = tid & 31;
  const int wave = tid >> 5;
  const int wm = wave >> 1, wn = wave & 1;

  v8f acc[2][4];
  #pragma unroll
  for (int i = 0; i < 2; i++)
    #pragma unroll
    for (int j = 0; j < 4; j++)
      #pragma unroll
      for (int r = 0; r < 8; r++) acc[i][j][r] = 0.0f;

  const int aRow = tid >> 1;            // 0..63
  const int aSeg = (tid & 1) * 16;      // half offset within row (0/16)
  const int bK   = tid >> 2;            // 0..31
  const int bGrp = (tid & 3) * 32;      // column group of 32

  const int am   = lane & 15;
  const int asel = (lane >> 4) * 8;     // A: lane-half K select (0 or 8)
  const int bn   = lane & 15;
  const int bsel = (lane >> 4) * 8;     // B: lane-half K select (0 or 8)

  const unsigned asOff = lds_off(&As[aRow * 32 + aSeg]);  // bytes

  for (int k0 = 0; k0 < K; k0 += 32) {
    // stage A (f16) 64x32 via async global->LDS, ASYNCcnt-tracked
    async_g2l_b128(asOff, A + (long)(m0 + aRow) * lda + k0 + aSeg);
    // stage B fp32 -> f16, transposed to [n][k]
    {
      const float* src = B + (long)(k0 + bK) * ldb + n0 + bGrp;
      if (k0 + 32 < K) __builtin_prefetch(src + 32L * ldb, 0, 1);
      #pragma unroll
      for (int j = 0; j < 32; j += 4) {
        float4 f = *(const float4*)(src + j);
        Bs[(bGrp + j + 0) * 32 + bK] = (_Float16)f.x;
        Bs[(bGrp + j + 1) * 32 + bK] = (_Float16)f.y;
        Bs[(bGrp + j + 2) * 32 + bK] = (_Float16)f.z;
        Bs[(bGrp + j + 3) * 32 + bK] = (_Float16)f.w;
      }
    }
    wait_async0();
    __syncthreads();

    F16x16 af[2], bf[4];
    #pragma unroll
    for (int mt = 0; mt < 2; mt++) {                 // A: lane=m, K split per half
      const _Float16* ap = &As[(wm * 32 + mt * 16 + am) * 32];
      af[mt].h[0] = *(const v8h*)(ap + asel);        // K 0-7  / 8-15
      af[mt].h[1] = *(const v8h*)(ap + 16 + asel);   // K 16-23 / 24-31
    }
    #pragma unroll
    for (int nt = 0; nt < 4; nt++) {                 // B: lane=n+16*khalf, K split per half as A
      const _Float16* bp = &Bs[(wn * 64 + nt * 16 + bn) * 32 + bsel];
      bf[nt].h[0] = *(const v8h*)(bp);
      bf[nt].h[1] = *(const v8h*)(bp + 16);
    }
    #pragma unroll
    for (int mt = 0; mt < 2; mt++)
      #pragma unroll
      for (int nt = 0; nt < 4; nt++)
        acc[mt][nt] = wmma_f16(af[mt], bf[nt], acc[mt][nt]);
    __syncthreads();
  }

  // epilogue: C layout -> lane=n (per half), VGPR r = row (r / 8+r)
  const int cn    = lane & 15;
  const int rbase = (lane >> 4) * 8;
  #pragma unroll
  for (int mt = 0; mt < 2; mt++)
    #pragma unroll
    for (int nt = 0; nt < 4; nt++) {
      const int col = n0 + wn * 64 + nt * 16 + cn;
      #pragma unroll
      for (int r = 0; r < 8; r++) {
        const long row = m0 + wm * 32 + mt * 16 + rbase + r;
        float v = acc[mt][nt][r];
        const long idx = row * ldc + col;
        if (ADD_RES) v += Res[idx];
        C[idx] = v;
      }
    }
}

// ---------------------------------------------------------------------------
// RMSNorm + f16 cast (one block per token)
// ---------------------------------------------------------------------------
__global__ __launch_bounds__(256)
void rmsnorm_cast(const float* __restrict__ x, const float* __restrict__ w,
                  _Float16* __restrict__ out)
{
  __shared__ float red[256];
  const long n = blockIdx.x;
  const float* xr = x + n * DIM;
  float s = 0.f;
  for (int d = threadIdx.x; d < DIM; d += 256) { float v = xr[d]; s += v * v; }
  red[threadIdx.x] = s; __syncthreads();
  for (int st = 128; st > 0; st >>= 1) {
    if (threadIdx.x < st) red[threadIdx.x] += red[threadIdx.x + st];
    __syncthreads();
  }
  const float rs = rsqrtf(red[0] * (1.0f / DIM) + 1e-6f);
  for (int d = threadIdx.x; d < DIM; d += 256)
    out[n * DIM + d] = (_Float16)(xr[d] * rs * w[d]);
}

// ---------------------------------------------------------------------------
// RoPE on q,k (fp32 in -> rotated f16 out), analytic tables
// ---------------------------------------------------------------------------
__global__ __launch_bounds__(256)
void rope_qk(const float* __restrict__ q32, const float* __restrict__ k32,
             _Float16* __restrict__ q16, _Float16* __restrict__ k16)
{
  const long i = (long)blockIdx.x * 256 + threadIdx.x;   // over NTOK*NH*32
  if (i >= (long)NTOK * NH * 32) return;
  const int  j = (int)(i & 31);
  const int  h = (int)((i >> 5) & 15);
  const long n = i >> 9;
  const int  s = (int)(n & (SEQ - 1));
  const float ang = (float)s / __powf(10000.f, (float)j * (1.f / 32.f));
  const float sn = __sinf(ang), cs = __cosf(ang);
  const long base = n * DIM + h * HD + j;
  float a = q32[base], b = q32[base + 32];
  q16[base]      = (_Float16)(a * cs - b * sn);
  q16[base + 32] = (_Float16)(b * cs + a * sn);
  a = k32[base];  b = k32[base + 32];
  k16[base]      = (_Float16)(a * cs - b * sn);
  k16[base + 32] = (_Float16)(b * cs + a * sn);
}

__global__ __launch_bounds__(256)
void cast_f32_f16(const float* __restrict__ in, _Float16* __restrict__ out, long n)
{
  const long i = (long)blockIdx.x * 256 + threadIdx.x;
  if (i < n) out[i] = (_Float16)in[i];
}

// ---------------------------------------------------------------------------
// Flash attention: one wave per 16-row Q tile per (b,h). Causal mask analytic.
// scores = q@k^T (no 1/sqrt(d) in reference). Online softmax in C-layout:
// rows per-VGPR, row reductions via shfl_xor within each 16-lane half.
// ---------------------------------------------------------------------------
__global__ __launch_bounds__(128)
void attn_flash(const _Float16* __restrict__ q, const _Float16* __restrict__ k,
                const _Float16* __restrict__ v, _Float16* __restrict__ out)
{
  __shared__ __align__(16) _Float16 smem[4 * 2560];   // per-wave: P 16x32 + Vt 64x32
  const int wave = threadIdx.x >> 5;
  const int lane = threadIdx.x & 31;
  _Float16* Pm = smem + wave * 2560;
  _Float16* Vt = smem + wave * 2560 + 512;

  const int tile = blockIdx.x * 4 + wave;   // 0..4095
  const int qt = tile & 127;
  const int h  = (tile >> 7) & 15;
  const int b  = tile >> 11;
  const long base_n = (long)b * SEQ;
  const int  col = h * HD;

  const int m     = lane & 15;
  const int asel  = (lane >> 4) * 8;
  const int bn    = lane & 15;
  const int bsel  = (lane >> 4) * 8;
  const int rbase = (lane >> 4) * 8;

  // Q fragments: A-layout, K-dim = HD = 64 -> two K-chunks of 32
  F16x16 qa[2];
  const _Float16* qp = q + (base_n + qt * 16 + m) * DIM + col;
  qa[0].h[0] = *(const v8h*)(qp + asel);
  qa[0].h[1] = *(const v8h*)(qp + 16 + asel);
  qa[1].h[0] = *(const v8h*)(qp + 32 + asel);
  qa[1].h[1] = *(const v8h*)(qp + 48 + asel);

  float rmax[8], rsum[8];
  v8f o[4];
  #pragma unroll
  for (int r = 0; r < 8; r++) { rmax[r] = -3.0e38f; rsum[r] = 0.f; }
  #pragma unroll
  for (int t = 0; t < 4; t++)
    #pragma unroll
    for (int r = 0; r < 8; r++) o[t][r] = 0.f;

  const int nkc = ((qt * 16 + 15) >> 5) + 1;          // causal chunk count (32 keys each)
  for (int kc = 0; kc < nkc; kc++) {
    const int kbase = kc * 32;

    // K fragments (B-layout of k^T): lane=(key, khalf), hd split per half like Q
    F16x16 kb[2][2];
    #pragma unroll
    for (int nt = 0; nt < 2; nt++) {
      const _Float16* kp = k + (base_n + kbase + nt * 16 + bn) * DIM + col;
      #pragma unroll
      for (int c = 0; c < 2; c++) {
        kb[nt][c].h[0] = *(const v8h*)(kp + c * 32 + bsel);
        kb[nt][c].h[1] = *(const v8h*)(kp + c * 32 + bsel + 16);
      }
    }
    v8f sc[2];
    #pragma unroll
    for (int nt = 0; nt < 2; nt++) {
      v8f z;
      #pragma unroll
      for (int r = 0; r < 8; r++) z[r] = 0.f;
      z = wmma_f16(qa[0], kb[nt][0], z);
      sc[nt] = wmma_f16(qa[1], kb[nt][1], z);
    }
    // causal mask (col 0 of mask is 0 already in reference)
    #pragma unroll
    for (int nt = 0; nt < 2; nt++) {
      const int key = kbase + nt * 16 + (lane & 15);
      #pragma unroll
      for (int r = 0; r < 8; r++) {
        const int qpos = qt * 16 + rbase + r;
        if (key > qpos) sc[nt][r] += -1.0e9f;
      }
    }
    // online softmax: row max / sum across 16-lane halves
    float alpha[8], psum[8];
    #pragma unroll
    for (int r = 0; r < 8; r++) {
      float mv = fmaxf(sc[0][r], sc[1][r]);
      #pragma unroll
      for (int off = 1; off <= 8; off <<= 1)
        mv = fmaxf(mv, __shfl_xor(mv, off, 32));
      const float mnew = fmaxf(rmax[r], mv);
      alpha[r] = __expf(rmax[r] - mnew);
      rmax[r] = mnew;
      psum[r] = 0.f;
    }
    #pragma unroll
    for (int nt = 0; nt < 2; nt++)
      #pragma unroll
      for (int r = 0; r < 8; r++) {
        const float p = __expf(sc[nt][r] - rmax[r]);
        psum[r] += p;
        Pm[(rbase + r) * 32 + nt * 16 + (lane & 15)] = (_Float16)p;
      }
    #pragma unroll
    for (int r = 0; r < 8; r++) {
      float sv = psum[r];
      #pragma unroll
      for (int off = 1; off <= 8; off <<= 1)
        sv += __shfl_xor(sv, off, 32);
      rsum[r] = rsum[r] * alpha[r] + sv;
    }
    #pragma unroll
    for (int t = 0; t < 4; t++)
      #pragma unroll
      for (int r = 0; r < 8; r++) o[t][r] *= alpha[r];

    // stage V transposed: Vt[hd][key], lane owns one key row
    {
      const _Float16* vp = v + (base_n + kbase + lane) * DIM + col;
      #pragma unroll 8
      for (int d = 0; d < HD; d++) Vt[d * 32 + lane] = vp[d];
    }
    asm volatile("s_wait_dscnt 0x0" ::: "memory");   // cross-lane LDS visibility

    // P in A-layout (K=32 keys), V in B-layout from Vt
    F16x16 pa;
    pa.h[0] = *(const v8h*)&Pm[m * 32 + asel];
    pa.h[1] = *(const v8h*)&Pm[m * 32 + 16 + asel];
    #pragma unroll
    for (int t = 0; t < 4; t++) {
      F16x16 vb;
      const _Float16* vbp = &Vt[(t * 16 + bn) * 32 + bsel];
      vb.h[0] = *(const v8h*)(vbp);
      vb.h[1] = *(const v8h*)(vbp + 16);
      o[t] = wmma_f16(pa, vb, o[t]);
    }
  }
  // normalize + store f16
  #pragma unroll
  for (int t = 0; t < 4; t++)
    #pragma unroll
    for (int r = 0; r < 8; r++) {
      const long row = base_n + qt * 16 + rbase + r;
      out[row * DIM + col + t * 16 + (lane & 15)] = (_Float16)(o[t][r] / rsum[r]);
    }
}

// ---------------------------------------------------------------------------
// MoE routing: gate logits + softmax + top2 (block per token), then a serial
// exact replica of the reference cumsum/capacity ranking (slots are unique, so
// dispatch is a gather).
// ---------------------------------------------------------------------------
__global__ __launch_bounds__(256)
void gate_top2(const _Float16* __restrict__ h2n, const float* __restrict__ gw,
               int* __restrict__ top1, int* __restrict__ top2,
               float* __restrict__ w1, float* __restrict__ w2)
{
  __shared__ float part[256];
  const long n = blockIdx.x;
  const int e = threadIdx.x & 15;
  const int g = threadIdx.x >> 4;
  float s = 0.f;
  for (int d = g * 64; d < g * 64 + 64; d++)
    s += (float)h2n[n * DIM + d] * gw[(long)d * NE + e];
  part[threadIdx.x] = s;
  __syncthreads();
  if (threadIdx.x == 0) {
    float lg[NE];
    for (int ee = 0; ee < NE; ee++) {
      float t = 0.f;
      for (int gg = 0; gg < 16; gg++) t += part[gg * 16 + ee];
      lg[ee] = t;
    }
    float mx = -3e38f;
    for (int ee = 0; ee < NE; ee++) mx = fmaxf(mx, lg[ee]);
    float p[NE], sum = 0.f;
    for (int ee = 0; ee < NE; ee++) { p[ee] = __expf(lg[ee] - mx); sum += p[ee]; }
    int t1 = 0;
    for (int ee = 1; ee < NE; ee++) if (p[ee] > p[t1]) t1 = ee;
    int t2 = (t1 == 0) ? 1 : 0;
    for (int ee = 0; ee < NE; ee++) if (ee != t1 && p[ee] > p[t2]) t2 = ee;
    top1[n] = t1; top2[n] = t2;
    w1[n] = p[t1] / sum; w2[n] = p[t2] / sum;
  }
}

__global__ void route_serial(const int* __restrict__ top1, const int* __restrict__ top2,
                             int* __restrict__ slot1, int* __restrict__ slot2,
                             int* __restrict__ src)
{
  if (threadIdx.x != 0 || blockIdx.x != 0) return;
  int c1t[NE], c1[NE], c2[NE];
  for (int e = 0; e < NE; e++) { c1t[e] = 0; c1[e] = 0; c2[e] = 0; }
  for (int i = 0; i < ECAP; i++) src[i] = -1;
  for (int t = 0; t < NTOK; t++) c1t[top1[t]]++;        // uncapped top1 totals
  for (int t = 0; t < NTOK; t++) {
    const int e1 = top1[t];
    const int r1 = c1[e1]++;
    if (r1 < CAPX) { slot1[t] = e1 * CAPX + r1; src[e1 * CAPX + r1] = t; }
    else slot1[t] = -1;
    const int e2 = top2[t];
    const int r2 = c2[e2]++ + c1t[e2];                  // rank2 offset = total top1 count
    if (r2 < CAPX) { slot2[t] = e2 * CAPX + r2; src[e2 * CAPX + r2] = t; }
    else slot2[t] = -1;
  }
}

__global__ __launch_bounds__(256)
void dispatch_gather(const int* __restrict__ src, const _Float16* __restrict__ h2n,
                     _Float16* __restrict__ disp)
{
  const int slot = blockIdx.x;
  const int t = src[slot];
  for (int d = threadIdx.x; d < DIM; d += 256)
    disp[(long)slot * DIM + d] = (t >= 0) ? h2n[(long)t * DIM + d] : (_Float16)0.f;
}

__global__ __launch_bounds__(256)
void swiglu_mul(const float* __restrict__ xg, const float* __restrict__ xu,
                _Float16* __restrict__ hm)
{
  const long i = (long)blockIdx.x * 256 + threadIdx.x;   // over ECAP*FFN1
  if (i >= (long)ECAP * FFN1) return;
  const long r = i >> 11;
  const int  f = (int)(i & (FFN1 - 1));
  const float x1 = xg[r * FFN2 + f];
  const float x2 = xg[r * FFN2 + FFN1 + f];
  const float u  = xu[i];
  const float sig = 1.f / (1.f + __expf(-x2));
  hm[i] = (_Float16)(x1 * (x2 * sig) * u);
}

__global__ __launch_bounds__(256)
void combine_out(const float* __restrict__ h1, const float* __restrict__ eo,
                 const int* __restrict__ slot1, const int* __restrict__ slot2,
                 const float* __restrict__ w1, const float* __restrict__ w2,
                 float* __restrict__ out)
{
  const long n = blockIdx.x;
  const int s1 = slot1[n], s2 = slot2[n];
  const float a = (s1 >= 0) ? w1[n] : 0.f;
  const float b = (s2 >= 0) ? w2[n] : 0.f;
  for (int d = threadIdx.x; d < DIM; d += 256) {
    float v = h1[n * DIM + d];
    if (s1 >= 0) v += a * eo[(long)s1 * DIM + d];
    if (s2 >= 0) v += b * eo[(long)s2 * DIM + d];
    out[n * DIM + d] = v;
  }
}

// ---------------------------------------------------------------------------
extern "C" void kernel_launch(void* const* d_in, const int* in_sizes, int n_in,
                              void* d_out, int out_size, void* d_ws, size_t ws_size,
                              hipStream_t stream) {
  const float* x      = (const float*)d_in[0];
  // d_in[1] attention_mask: causal -1e9, reproduced analytically
  const float* ln1_w  = (const float*)d_in[2];
  const float* ln2_w  = (const float*)d_in[3];
  const float* Wq     = (const float*)d_in[4];
  const float* Wk     = (const float*)d_in[5];
  const float* Wv     = (const float*)d_in[6];
  const float* Wo     = (const float*)d_in[7];
  const float* gate_w = (const float*)d_in[8];
  const float* wig    = (const float*)d_in[9];   // [E,1024,4096]
  const float* wiu    = (const float*)d_in[10];  // [E,1024,2048]
  const float* woe    = (const float*)d_in[11];  // [E,2048,1024]
  float* out = (float*)d_out;

  char* ws = (char*)d_ws;
  size_t off = 0;
  auto alloc = [&](size_t bytes) { size_t o = off; off = (off + bytes + 255) & ~(size_t)255; return o; };

  _Float16* xn16   = (_Float16*)(ws + alloc((size_t)NTOK * DIM * 2));
  float*    q32    = (float*)   (ws + alloc((size_t)NTOK * DIM * 4));
  float*    k32    = (float*)   (ws + alloc((size_t)NTOK * DIM * 4));
  float*    v32    = (float*)   (ws + alloc((size_t)NTOK * DIM * 4));
  _Float16* q16    = (_Float16*)(ws + alloc((size_t)NTOK * DIM * 2));
  _Float16* k16    = (_Float16*)(ws + alloc((size_t)NTOK * DIM * 2));
  _Float16* v16    = (_Float16*)(ws + alloc((size_t)NTOK * DIM * 2));
  _Float16* attn16 = (_Float16*)(ws + alloc((size_t)NTOK * DIM * 2));
  float*    h1     = (float*)   (ws + alloc((size_t)NTOK * DIM * 4));
  _Float16* h2n16  = (_Float16*)(ws + alloc((size_t)NTOK * DIM * 2));
  int*      top1   = (int*)     (ws + alloc((size_t)NTOK * 4));
  int*      top2   = (int*)     (ws + alloc((size_t)NTOK * 4));
  float*    w1     = (float*)   (ws + alloc((size_t)NTOK * 4));
  float*    w2     = (float*)   (ws + alloc((size_t)NTOK * 4));
  int*      slot1  = (int*)     (ws + alloc((size_t)NTOK * 4));
  int*      slot2  = (int*)     (ws + alloc((size_t)NTOK * 4));
  int*      src    = (int*)     (ws + alloc((size_t)ECAP * 4));
  _Float16* disp16 = (_Float16*)(ws + alloc((size_t)ECAP * DIM * 2));
  float*    xg32   = (float*)   (ws + alloc((size_t)ECAP * FFN2 * 4));
  float*    xu32   = (float*)   (ws + alloc((size_t)ECAP * FFN1 * 4));
  _Float16* hm16   = (_Float16*)(ws + alloc((size_t)ECAP * FFN1 * 2));
  float*    eo32   = (float*)   (ws + alloc((size_t)ECAP * DIM * 4));
  (void)ws_size; (void)in_sizes; (void)n_in; (void)out_size;

  // 1. rmsnorm1
  rmsnorm_cast<<<NTOK, 256, 0, stream>>>(x, ln1_w, xn16);
  // 2. Q/K/V projections (WMMA)
  dim3 gProj(DIM / 128, NTOK / 64, 1);
  gemm_wmma<false><<<gProj, 128, 0, stream>>>(xn16, DIM, 0, Wq, DIM, 0, q32, DIM, 0, nullptr, DIM);
  gemm_wmma<false><<<gProj, 128, 0, stream>>>(xn16, DIM, 0, Wk, DIM, 0, k32, DIM, 0, nullptr, DIM);
  gemm_wmma<false><<<gProj, 128, 0, stream>>>(xn16, DIM, 0, Wv, DIM, 0, v32, DIM, 0, nullptr, DIM);
  // 3. RoPE + casts
  rope_qk<<<((long)NTOK * NH * 32 + 255) / 256, 256, 0, stream>>>(q32, k32, q16, k16);
  cast_f32_f16<<<((long)NTOK * DIM + 255) / 256, 256, 0, stream>>>(v32, v16, (long)NTOK * DIM);
  // 4. flash attention (1 wave per 16-row q tile; 4 waves / block)
  attn_flash<<<(2 * NH * (SEQ / 16)) / 4, 128, 0, stream>>>(q16, k16, v16, attn16);
  // 5. output projection + residual
  gemm_wmma<true><<<gProj, 128, 0, stream>>>(attn16, DIM, 0, Wo, DIM, 0, h1, DIM, 0, x, DIM);
  // 6. rmsnorm2
  rmsnorm_cast<<<NTOK, 256, 0, stream>>>(h1, ln2_w, h2n16);
  // 7-8. routing
  gate_top2<<<NTOK, 256, 0, stream>>>(h2n16, gate_w, top1, top2, w1, w2);
  route_serial<<<1, 1, 0, stream>>>(top1, top2, slot1, slot2, src);
  // 9. dispatch gather
  dispatch_gather<<<ECAP, 256, 0, stream>>>(src, h2n16, disp16);
  // 10-11. expert GEMMs (batched over z = expert)
  dim3 gG(FFN2 / 128, CAPX / 64, NE);
  gemm_wmma<false><<<gG, 128, 0, stream>>>(disp16, DIM, (long)CAPX * DIM,
                                           wig, FFN2, (long)DIM * FFN2,
                                           xg32, FFN2, (long)CAPX * FFN2, nullptr, DIM);
  dim3 gU(FFN1 / 128, CAPX / 64, NE);
  gemm_wmma<false><<<gU, 128, 0, stream>>>(disp16, DIM, (long)CAPX * DIM,
                                           wiu, FFN1, (long)DIM * FFN1,
                                           xu32, FFN1, (long)CAPX * FFN1, nullptr, DIM);
  // 12. swiglu * up
  swiglu_mul<<<((long)ECAP * FFN1 + 255) / 256, 256, 0, stream>>>(xg32, xu32, hm16);
  // 13. down projection
  dim3 gO(DIM / 128, CAPX / 64, NE);
  gemm_wmma<false><<<gO, 128, 0, stream>>>(hm16, FFN1, (long)CAPX * FFN1,
                                           woe, DIM, (long)FFN1 * DIM,
                                           eo32, DIM, (long)CAPX * DIM, nullptr, FFN1);
  // 14. weighted combine + residual
  combine_out<<<NTOK, 256, 0, stream>>>(h1, eo32, slot1, slot2, w1, w2, out);
}
